// DeltaNet_70317204570435
// MI455X (gfx1250) — matrix operand order repaired and checked by reading, not compiled, not run on hardware
//
#include <hip/hip_runtime.h>
#include <hip/hip_bf16.h>

typedef __bf16 bf16_t;
typedef __attribute__((ext_vector_type(16))) __bf16 bf16x16;
typedef __attribute__((ext_vector_type(8)))  __bf16 bf16x8;
typedef __attribute__((ext_vector_type(8)))  float  f32x8;

#define H_DIM 1024
#define NHEAD 32
#define HD    32
#define SEQ_T 1024

// ---------------- f32 -> bf16 conversion ----------------
__global__ __launch_bounds__(256) void cvt_f32_to_bf16(const float* __restrict__ in,
                                                       bf16_t* __restrict__ out, int n) {
  int i = blockIdx.x * 256 + threadIdx.x;
  if (i < n) out[i] = (bf16_t)in[i];
}

// ---------------- fragment loaders (CDNA5 16x16x32 bf16 layouts) ----------------
__device__ __forceinline__ bf16x16 load_frag_a(const bf16_t* ap) {
  // A 16x32: lane holds row m=lane&15, 16 contiguous K (ap pre-offset by hs*16)
  union { bf16x16 v; bf16x8 h[2]; } u;
  u.h[0] = *(const bf16x8*)(ap);
  u.h[1] = *(const bf16x8*)(ap + 8);
  return u.v;
}
__device__ __forceinline__ bf16x16 load_frag_b(const bf16_t* bp) {
  // B 32x16: lane holds column n=lane&15, 16 contiguous K (bp pre-offset by hs*16)
  union { bf16x16 v; bf16x8 h[2]; } u;
  u.h[0] = *(const bf16x8*)(bp);
  u.h[1] = *(const bf16x8*)(bp + 8);
  return u.v;
}
__device__ __forceinline__ f32x8 wmma_bf16(bf16x16 a, bf16x16 b, f32x8 c) {
  return __builtin_amdgcn_wmma_f32_16x16x32_bf16(false, a, false, b, (short)0, c, false, false);
}

// ---------------- bf16 WMMA GEMM: Y[M,N] = A[M,KC] @ W[N,KC]^T (+bias[n]) (+resid[m,n]) ----------------
// A row-major (M,KC) bf16; W row-major (N,KC) bf16. One wave: 16(M) x 64(N) strip; block = 8 waves.
// Explicit two-buffer ping-pong, peeled epilogue, partial unroll: WMMAs consume fragments loaded
// half an iteration earlier -> staggered loadcnt waits, no rotation copies.
template <int KC>
__global__ __launch_bounds__(256) void gemm_bf16_wmma(const bf16_t* __restrict__ A,
                                                      const bf16_t* __restrict__ W,
                                                      float* __restrict__ Y,
                                                      const float* __restrict__ bias,
                                                      const float* __restrict__ resid,
                                                      int M, int N) {
  const int wave = threadIdx.x >> 5;
  const int lane = threadIdx.x & 31;
  const int row0 = blockIdx.y * 128 + wave * 16;
  const int col0 = blockIdx.x * 64;
  if (row0 >= M) return;
  const int hs  = lane >> 4;
  const int r16 = lane & 15;

  const bf16_t* ap  = A + (size_t)(row0 + r16) * KC + hs * 16;
  const bf16_t* bp0 = W + (size_t)(col0 + r16) * KC + hs * 16;
  const bf16_t* bp1 = W + (size_t)(col0 + 16 + r16) * KC + hs * 16;
  const bf16_t* bp2 = W + (size_t)(col0 + 32 + r16) * KC + hs * 16;
  const bf16_t* bp3 = W + (size_t)(col0 + 48 + r16) * KC + hs * 16;

  f32x8 acc[4] = {};
  // ping buffer: slab 0
  bf16x16 aA  = load_frag_a(ap);
  bf16x16 bA0 = load_frag_b(bp0), bA1 = load_frag_b(bp1),
          bA2 = load_frag_b(bp2), bA3 = load_frag_b(bp3);
  bf16x16 aB, bB0, bB1, bB2, bB3;

#pragma unroll 2
  for (int kb = 0; kb < KC - 64; kb += 64) {
    // pong loads (slab kb+32) issued before ping WMMAs
    aB  = load_frag_a(ap + kb + 32);
    bB0 = load_frag_b(bp0 + kb + 32);
    bB1 = load_frag_b(bp1 + kb + 32);
    bB2 = load_frag_b(bp2 + kb + 32);
    bB3 = load_frag_b(bp3 + kb + 32);
    __builtin_prefetch(ap + kb + 160, 0, 1);
    acc[0] = wmma_bf16(aA, bA0, acc[0]);
    acc[1] = wmma_bf16(aA, bA1, acc[1]);
    acc[2] = wmma_bf16(aA, bA2, acc[2]);
    acc[3] = wmma_bf16(aA, bA3, acc[3]);
    // ping loads (slab kb+64) issued before pong WMMAs; loop-carried into next iter
    aA  = load_frag_a(ap + kb + 64);
    bA0 = load_frag_b(bp0 + kb + 64);
    bA1 = load_frag_b(bp1 + kb + 64);
    bA2 = load_frag_b(bp2 + kb + 64);
    bA3 = load_frag_b(bp3 + kb + 64);
    acc[0] = wmma_bf16(aB, bB0, acc[0]);
    acc[1] = wmma_bf16(aB, bB1, acc[1]);
    acc[2] = wmma_bf16(aB, bB2, acc[2]);
    acc[3] = wmma_bf16(aB, bB3, acc[3]);
  }
  // epilogue: slabs KC-64 (in ping regs) and KC-32
  aB  = load_frag_a(ap + KC - 32);
  bB0 = load_frag_b(bp0 + KC - 32);
  bB1 = load_frag_b(bp1 + KC - 32);
  bB2 = load_frag_b(bp2 + KC - 32);
  bB3 = load_frag_b(bp3 + KC - 32);
  acc[0] = wmma_bf16(aA, bA0, acc[0]);
  acc[1] = wmma_bf16(aA, bA1, acc[1]);
  acc[2] = wmma_bf16(aA, bA2, acc[2]);
  acc[3] = wmma_bf16(aA, bA3, acc[3]);
  acc[0] = wmma_bf16(aB, bB0, acc[0]);
  acc[1] = wmma_bf16(aB, bB1, acc[1]);
  acc[2] = wmma_bf16(aB, bB2, acc[2]);
  acc[3] = wmma_bf16(aB, bB3, acc[3]);

  // C/D layout: VGPR r -> row (r + 8*hs), column r16 (+ tile offset)
#pragma unroll
  for (int nt = 0; nt < 4; ++nt) {
    int n = col0 + nt * 16 + r16;
    float badd = bias ? bias[n] : 0.0f;
#pragma unroll
    for (int r = 0; r < 8; ++r) {
      int mrow = row0 + r + hs * 8;
      float val = acc[nt][r] + badd;
      if (resid) val += resid[(size_t)mrow * N + n];
      Y[(size_t)mrow * N + n] = val;
    }
  }
}

// ---------------- RoPE + phi (elu+1) on q,k ; sigmoid/clip beta ; softmax(M=2) mix ----------------
__device__ __forceinline__ float phi_elu1(float x) { return x > 0.0f ? x + 1.0f : __expf(x); }

__global__ __launch_bounds__(256) void rope_phi_act(float* __restrict__ q, float* __restrict__ k,
                                                    float* __restrict__ beta, float* __restrict__ mix) {
  int row = blockIdx.x;              // b*t index
  int t = row & (SEQ_T - 1);
  int tid = threadIdx.x;
  size_t base = (size_t)row * H_DIM;
  for (int p = tid; p < 1024; p += 256) {
    float* ptr = (p < 512) ? q : k;
    int pp = p & 511;
    int h = pp >> 4, j = pp & 15;
    float inv = __powf(10000.0f, -(float)j * (1.0f / 16.0f));
    float sn, cs;
    __sincosf((float)t * inv, &sn, &cs);
    int i1 = h * HD + j, i2 = i1 + 16;
    float u1 = ptr[base + i1], u2 = ptr[base + i2];
    ptr[base + i1] = phi_elu1(u1 * cs - u2 * sn);
    ptr[base + i2] = phi_elu1(u1 * sn + u2 * cs);
  }
  if (tid < 64) {
    size_t bb = (size_t)row * 64 + tid;
    float s = 1.0f / (1.0f + __expf(-beta[bb]));
    beta[bb] = fminf(fmaxf(s, 0.85f), 0.9995f);
  } else if (tid < 96) {
    size_t bb = (size_t)row * 64 + (size_t)(tid - 64) * 2;
    float a = mix[bb], b = mix[bb + 1];
    float mx = fmaxf(a, b);
    float ea = __expf(a - mx), eb = __expf(b - mx);
    float s = 1.0f / (ea + eb);
    mix[bb] = ea * s;
    mix[bb + 1] = eb * s;
  }
}

// ---------------- sequential decay scan ----------------
// One block per (b,h). thread tid: m = tid>>7, c = tid&31, dg = (tid>>5)&3 ;
// owns S[m][dg*8+i][c] in registers. Inputs double-buffered in LDS, global loads
// prefetched one timestep ahead into registers -> 2 barriers/step.
__global__ __launch_bounds__(256) void decay_scan(const float* __restrict__ phiq,
                                                  const float* __restrict__ phik,
                                                  const float* __restrict__ vv,
                                                  const float* __restrict__ beta,
                                                  const float* __restrict__ mix,
                                                  float* __restrict__ y) {
  const int bh = blockIdx.x;
  const int bi = bh / NHEAD, h = bh % NHEAD;
  const int tid = threadIdx.x;
  const int m  = tid >> 7;
  const int c  = tid & 31;
  const int dg = (tid >> 5) & 3;

  // input slot layout: [0..31]=pq, [32..63]=pk, [64..95]=v, [96..97]=beta, [98..99]=mix
  __shared__ float s_in[2][100];
  __shared__ float s_K[2][32];
  __shared__ float s_red[256];

  const bool has_in = tid < 100;
  auto fetch = [&](int t) -> float {
    size_t row  = (size_t)(bi * SEQ_T + t);
    size_t base = row * H_DIM + (size_t)h * HD;
    size_t bm   = row * 64 + (size_t)h * 2;
    if (tid < 32)  return phiq[base + tid];
    if (tid < 64)  return phik[base + tid - 32];
    if (tid < 96)  return vv[base + tid - 64];
    if (tid < 98)  return beta[bm + (tid - 96)];
    return mix[bm + (tid - 98)];
  };

  float S[8];
#pragma unroll
  for (int i = 0; i < 8; ++i) S[i] = 0.0f;
  if (tid < 64) s_K[tid >> 5][tid & 31] = 0.0f;

  float r_in = has_in ? fetch(0) : 0.0f;

  for (int t = 0; t < SEQ_T; ++t) {
    const int p = t & 1;
    if (has_in) s_in[p][tid] = r_in;
    if (t + 1 < SEQ_T && has_in) r_in = fetch(t + 1);  // overlaps step-t compute
    __syncthreads();

    const float* in = s_in[p];
    float bt = in[96 + m];
    if (tid < 64) {
      int km = tid >> 5, kd = tid & 31;
      s_K[km][kd] = s_K[km][kd] * in[96 + km] + in[32 + kd];
    }
    float vc = in[64 + c];
    float pn = 0.0f;
#pragma unroll
    for (int i = 0; i < 8; ++i) {
      int d = dg * 8 + i;
      S[i] = S[i] * bt + in[32 + d] * vc;
      pn += in[d] * S[i];
    }
    s_red[tid] = pn;
    __syncthreads();

    if (tid < 32) {
      float den0 = 1e-6f, den1 = 1e-6f;
#pragma unroll
      for (int d = 0; d < 32; ++d) {
        den0 += in[d] * s_K[0][d];
        den1 += in[d] * s_K[1][d];
      }
      float num0 = s_red[c] + s_red[32 + c] + s_red[64 + c] + s_red[96 + c];
      float num1 = s_red[128 + c] + s_red[160 + c] + s_red[192 + c] + s_red[224 + c];
      size_t ybase = (size_t)(bi * SEQ_T + t) * H_DIM + (size_t)h * HD;
      y[ybase + c] = in[98] * num0 / den0 + in[99] * num1 / den1;
    }
  }
}

// ---------------- per-row LayerNorm ----------------
__global__ __launch_bounds__(256) void layernorm_row(const float* __restrict__ z,
                                                     const float* __restrict__ g,
                                                     const float* __restrict__ b,
                                                     float* __restrict__ out) {
  int row = blockIdx.x, tid = threadIdx.x;
  __shared__ float red[256];
  const float* zr = z + (size_t)row * H_DIM;
  float s = 0.0f;
  for (int i = tid; i < H_DIM; i += 256) s += zr[i];
  red[tid] = s;
  __syncthreads();
  for (int off = 128; off > 0; off >>= 1) {
    if (tid < off) red[tid] += red[tid + off];
    __syncthreads();
  }
  float mu = red[0] * (1.0f / H_DIM);
  __syncthreads();
  float vs = 0.0f;
  for (int i = tid; i < H_DIM; i += 256) { float d = zr[i] - mu; vs += d * d; }
  red[tid] = vs;
  __syncthreads();
  for (int off = 128; off > 0; off >>= 1) {
    if (tid < off) red[tid] += red[tid + off];
    __syncthreads();
  }
  float rstd = rsqrtf(red[0] * (1.0f / H_DIM) + 1e-5f);
  for (int i = tid; i < H_DIM; i += 256)
    out[(size_t)row * H_DIM + i] = (zr[i] - mu) * rstd * g[i] + b[i];
}

// ---------------- launch ----------------
extern "C" void kernel_launch(void* const* d_in, const int* in_sizes, int n_in,
                              void* d_out, int out_size, void* d_ws, size_t ws_size,
                              hipStream_t stream) {
  (void)n_in; (void)out_size; (void)ws_size;
  const float* x     = (const float*)d_in[0];
  const float* Wq    = (const float*)d_in[1];
  const float* Wk    = (const float*)d_in[2];
  const float* Wv    = (const float*)d_in[3];
  const float* Wbeta = (const float*)d_in[4];
  const float* bbeta = (const float*)d_in[5];
  const float* Wmix  = (const float*)d_in[6];
  const float* bmix  = (const float*)d_in[7];
  const float* Wout  = (const float*)d_in[8];
  const float* bout  = (const float*)d_in[9];
  const float* ln_g  = (const float*)d_in[10];
  const float* ln_b  = (const float*)d_in[11];

  const int Mr = in_sizes[0] / H_DIM;   // b * t = 4096
  const int B  = Mr / SEQ_T;            // 4

  char* ws = (char*)d_ws;
  size_t off = 0;
  auto take = [&](size_t bytes) -> char* {
    off = (off + 255) & ~(size_t)255;
    char* p = ws + off;
    off += bytes;
    return p;
  };
  bf16_t* xb     = (bf16_t*)take((size_t)Mr * H_DIM * 2);
  bf16_t* wqb    = (bf16_t*)take((size_t)H_DIM * H_DIM * 2);
  bf16_t* wkb    = (bf16_t*)take((size_t)H_DIM * H_DIM * 2);
  bf16_t* wvb    = (bf16_t*)take((size_t)H_DIM * H_DIM * 2);
  bf16_t* woutb  = (bf16_t*)take((size_t)H_DIM * H_DIM * 2);
  bf16_t* wbetab = (bf16_t*)take((size_t)64 * H_DIM * 2);
  bf16_t* wmixb  = (bf16_t*)take((size_t)64 * H_DIM * 2);
  float*  qf     = (float*)take((size_t)Mr * H_DIM * 4);
  float*  kf     = (float*)take((size_t)Mr * H_DIM * 4);
  float*  vf     = (float*)take((size_t)Mr * H_DIM * 4);
  float*  betaf  = (float*)take((size_t)Mr * 64 * 4);
  float*  mixf   = (float*)take((size_t)Mr * 64 * 4);
  float*  yf     = (float*)take((size_t)Mr * H_DIM * 4);
  bf16_t* yb     = (bf16_t*)kf;   // alias: k no longer needed after scan
  float*  zf     = qf;            // alias: q no longer needed after scan

  dim3 blk(256);
  const int nX = Mr * H_DIM, nW = H_DIM * H_DIM, nSmall = 64 * H_DIM;
  cvt_f32_to_bf16<<<(nX + 255) / 256, blk, 0, stream>>>(x, xb, nX);
  cvt_f32_to_bf16<<<(nW + 255) / 256, blk, 0, stream>>>(Wq, wqb, nW);
  cvt_f32_to_bf16<<<(nW + 255) / 256, blk, 0, stream>>>(Wk, wkb, nW);
  cvt_f32_to_bf16<<<(nW + 255) / 256, blk, 0, stream>>>(Wv, wvb, nW);
  cvt_f32_to_bf16<<<(nW + 255) / 256, blk, 0, stream>>>(Wout, woutb, nW);
  cvt_f32_to_bf16<<<(nSmall + 255) / 256, blk, 0, stream>>>(Wbeta, wbetab, nSmall);
  cvt_f32_to_bf16<<<(nSmall + 255) / 256, blk, 0, stream>>>(Wmix, wmixb, nSmall);

  dim3 gbig(H_DIM / 64, Mr / 128);
  gemm_bf16_wmma<H_DIM><<<gbig, blk, 0, stream>>>(xb, wqb, qf, nullptr, nullptr, Mr, H_DIM);
  gemm_bf16_wmma<H_DIM><<<gbig, blk, 0, stream>>>(xb, wkb, kf, nullptr, nullptr, Mr, H_DIM);
  gemm_bf16_wmma<H_DIM><<<gbig, blk, 0, stream>>>(xb, wvb, vf, nullptr, nullptr, Mr, H_DIM);
  dim3 gsmall(1, Mr / 128);
  gemm_bf16_wmma<H_DIM><<<gsmall, blk, 0, stream>>>(xb, wbetab, betaf, bbeta, nullptr, Mr, 64);
  gemm_bf16_wmma<H_DIM><<<gsmall, blk, 0, stream>>>(xb, wmixb, mixf, bmix, nullptr, Mr, 64);

  rope_phi_act<<<Mr, blk, 0, stream>>>(qf, kf, betaf, mixf);

  decay_scan<<<B * NHEAD, blk, 0, stream>>>(qf, kf, vf, betaf, mixf, yf);

  cvt_f32_to_bf16<<<(nX + 255) / 256, blk, 0, stream>>>(yf, yb, nX);
  gemm_bf16_wmma<H_DIM><<<gbig, blk, 0, stream>>>(yb, woutb, zf, bout, x, Mr, H_DIM);

  layernorm_row<<<Mr, blk, 0, stream>>>(zf, ln_g, ln_b, (float*)d_out);
}
